// SpatialMessagePattern_90984587199114
// MI455X (gfx1250) — matrix operand order repaired and checked by reading, not compiled, not run on hardware
//
#include <hip/hip_runtime.h>
#include <hip/hip_bf16.h>

// ---------------------------------------------------------------------------
// Watermark pattern generator for MI455X (gfx1250, wave32, WMMA).
// GEMM chain runs on v_wmma_f32_16x16x32_f16; epilogue fuses bias + act.
// K-loop unrolled 2x with global_prefetch of the k+2 fragments.
// ---------------------------------------------------------------------------

typedef _Float16 h16;
typedef __attribute__((ext_vector_type(16))) _Float16 v16h;
typedef __attribute__((ext_vector_type(8)))  _Float16 v8h;
typedef __attribute__((ext_vector_type(8)))  float    v8f;

#define BATCH 4096
#define PP    64

// ---------------- f32 -> f16 convert (row-major) ----------------
__global__ __launch_bounds__(256) void k_cvt_f16(const float* __restrict__ src,
                                                 h16* __restrict__ dst, int n) {
  int i = blockIdx.x * 256 + threadIdx.x;
  if (i < n) dst[i] = (h16)src[i];
}

// ---------------- W[K,N] f32 -> Wt[N,K] f16 (transpose) ----------------
__global__ __launch_bounds__(256) void k_transpose_f16(const float* __restrict__ W,
                                                       h16* __restrict__ Wt,
                                                       int K, int N) {
  int i = blockIdx.x * 256 + threadIdx.x;
  if (i < K * N) {
    int k = i / N, n = i - k * N;
    Wt[n * K + k] = (h16)W[i];
  }
}

// Load a 16-f16 WMMA fragment as two contiguous 16-byte chunks.
// A-frag: gap = 16 elements (K = 8h+0..7 then K = 16+8h+0..7)
// B-frag: gap = 16 elements (same K order as A-frag)
__device__ __forceinline__ v16h ld_frag(const h16* p, int gap) {
  v8h lo = *(const v8h*)(p);
  v8h hi = *(const v8h*)(p + gap);
  return __builtin_shufflevector(lo, hi, 0, 1, 2, 3, 4, 5, 6, 7,
                                         8, 9, 10, 11, 12, 13, 14, 15);
}

// ---------------- WMMA GEMM: out = act(A[M,K] @ Wt[N,K]^T + bias) ---------
// One wave per block. Wave computes a 32(M) x 64(N) tile: 2x4 accumulators,
// 8 v_wmma_f32_16x16x32_f16 per K-step, K-loop unrolled by 2 (K%64 == 0).
// act: 0 = relu, 1 = tanh. outH (f16) or outF (f32), the other nullptr.
__global__ __launch_bounds__(32) void k_gemm_wmma(
    const h16* __restrict__ A, const h16* __restrict__ Bt,
    const float* __restrict__ bias, h16* __restrict__ outH,
    float* __restrict__ outF, int M, int N, int K, int act) {
  const int lane = threadIdx.x;
  const int lm = lane & 15;       // M (A), N (B/C) index within tile
  const int h  = lane >> 4;       // K-half selector (A/B), M-half (C)
  const int nBase = blockIdx.x * 64;
  const int mBase = blockIdx.y * 32;

  v8f acc[2][4];
#pragma unroll
  for (int mt = 0; mt < 2; ++mt)
#pragma unroll
    for (int nt = 0; nt < 4; ++nt) acc[mt][nt] = (v8f){};

  const h16* aRow[2];
  aRow[0] = A + (size_t)(mBase + lm) * K + 8 * h;
  aRow[1] = A + (size_t)(mBase + 16 + lm) * K + 8 * h;
  const h16* bCol[4];
#pragma unroll
  for (int nt = 0; nt < 4; ++nt)
    bCol[nt] = Bt + (size_t)(nBase + nt * 16 + lm) * K + 8 * h;

  const int ksteps = K >> 5;      // always even for our shapes
  for (int kk = 0; kk < ksteps; kk += 2) {
    const int ko = kk << 5;
    // Prefetch the k+2 fragments toward this WGP while WMMAs run.
#pragma unroll
    for (int mt = 0; mt < 2; ++mt)
      __builtin_prefetch(aRow[mt] + ko + 64, 0, 3);
#pragma unroll
    for (int nt = 0; nt < 4; ++nt)
      __builtin_prefetch(bCol[nt] + ko + 64, 0, 3);

#pragma unroll
    for (int u = 0; u < 2; ++u) {
      const int kou = ko + (u << 5);
      v16h a[2], b[4];
#pragma unroll
      for (int mt = 0; mt < 2; ++mt) a[mt] = ld_frag(aRow[mt] + kou, 16);
#pragma unroll
      for (int nt = 0; nt < 4; ++nt) b[nt] = ld_frag(bCol[nt] + kou, 16);
#pragma unroll
      for (int mt = 0; mt < 2; ++mt)
#pragma unroll
        for (int nt = 0; nt < 4; ++nt)
          acc[mt][nt] = __builtin_amdgcn_wmma_f32_16x16x32_f16(
              false, a[mt], false, b[nt], (short)0, acc[mt][nt], false, false);
    }
  }

  // Epilogue: bias + activation, store. C layout: elem r -> M = r + 8h, N = lm.
#pragma unroll
  for (int mt = 0; mt < 2; ++mt)
#pragma unroll
    for (int nt = 0; nt < 4; ++nt) {
      const int col = nBase + nt * 16 + lm;
      const float bv = bias[col];
#pragma unroll
      for (int r = 0; r < 8; ++r) {
        const int row = mBase + mt * 16 + h * 8 + r;
        float v = acc[mt][nt][r] + bv;
        v = act ? tanhf(v) : fmaxf(v, 0.0f);
        if (outH) outH[(size_t)row * N + col] = (h16)v;
        else      outF[(size_t)row * N + col] = v;
      }
    }
}

// -------- assemble 3 channels + separable 5x5 Gaussian blur, per image ----
__global__ __launch_bounds__(256) void k_assemble_blur(
    const float* __restrict__ spatial,  // [B,64,64]
    const float* __restrict__ freqlo,   // [B,16,16]
    const float* __restrict__ msg,      // [B,64] original f32 message
    float* __restrict__ out) {          // [B,3,64,64]
  __shared__ float t0[64 * 66];
  __shared__ float t1[64 * 66];
  __shared__ float flo[256];
  __shared__ float mb[64];

  const int b = blockIdx.x;
  const int tid = threadIdx.x;
  // Normalized 5-tap Gaussian, sigma = 1.
  const float w0 = 0.4026199470f, w1 = 0.2442013420f, w2 = 0.0544886850f;
  const float wt[5] = {w2, w1, w0, w1, w2};

  if (tid < 256) flo[tid] = freqlo[b * 256 + tid];
  if (tid < 64)  mb[tid]  = msg[b * 64 + tid];
  __syncthreads();

  for (int ch = 0; ch < 3; ++ch) {
    // Build channel image into t0.
    for (int i = tid; i < 4096; i += 256) {
      const int p = i >> 6, q = i & 63;
      float v;
      if (ch == 0) {
        v = spatial[(size_t)b * 4096 + i];
      } else if (ch == 1) {
        // bilinear upsample 16 -> 64, align_corners=True
        const float pf = p * (15.0f / 63.0f);
        const float qf = q * (15.0f / 63.0f);
        const int pl = (int)pf, ql = (int)qf;
        const int ph = min(pl + 1, 15), qh = min(ql + 1, 15);
        const float wp = pf - (float)pl, wq = qf - (float)ql;
        const float v00 = flo[pl * 16 + ql], v01 = flo[pl * 16 + qh];
        const float v10 = flo[ph * 16 + ql], v11 = flo[ph * 16 + qh];
        v = (1.0f - wp) * ((1.0f - wq) * v00 + wq * v01) +
            wp * ((1.0f - wq) * v10 + wq * v11);
      } else {
        v = mb[((p >> 3) << 3) + (q >> 3)];  // 8x8 block pattern
      }
      t0[p * 66 + q] = v;
    }
    __syncthreads();

    // Horizontal pass (zero padded).
    for (int i = tid; i < 4096; i += 256) {
      const int p = i >> 6, q = i & 63;
      float s = 0.0f;
#pragma unroll
      for (int d = 0; d < 5; ++d) {
        const int qq = q + d - 2;
        if (qq >= 0 && qq < 64) s += wt[d] * t0[p * 66 + qq];
      }
      t1[p * 66 + q] = s;
    }
    __syncthreads();

    // Vertical pass + store.
    for (int i = tid; i < 4096; i += 256) {
      const int p = i >> 6, q = i & 63;
      float s = 0.0f;
#pragma unroll
      for (int d = 0; d < 5; ++d) {
        const int pp = p + d - 2;
        if (pp >= 0 && pp < 64) s += wt[d] * t1[pp * 66 + q];
      }
      out[(((size_t)b * 3 + ch) << 12) + i] = s;
    }
    __syncthreads();
  }
}

// ---------------------------------------------------------------------------
extern "C" void kernel_launch(void* const* d_in, const int* in_sizes, int n_in,
                              void* d_out, int out_size, void* d_ws, size_t ws_size,
                              hipStream_t stream) {
  (void)in_sizes; (void)n_in; (void)out_size; (void)ws_size;

  const float* msg = (const float*)d_in[0];
  const float* sW1 = (const float*)d_in[1];
  const float* sb1 = (const float*)d_in[2];
  const float* sW2 = (const float*)d_in[3];
  const float* sb2 = (const float*)d_in[4];
  const float* sW3 = (const float*)d_in[5];
  const float* sb3 = (const float*)d_in[6];
  const float* fW1 = (const float*)d_in[7];
  const float* fb1 = (const float*)d_in[8];
  const float* fW2 = (const float*)d_in[9];
  const float* fb2 = (const float*)d_in[10];
  float* out = (float*)d_out;

  // Workspace carve-up (256-byte aligned).
  char* ws = (char*)d_ws;
  size_t off = 0;
  auto carve = [&](size_t bytes) -> void* {
    void* p = ws + off;
    off += (bytes + 255) & ~(size_t)255;
    return p;
  };
  h16* msgH = (h16*)carve((size_t)BATCH * 64 * 2);
  h16* wt1  = (h16*)carve((size_t)64 * 256 * 2);    // [256][64]
  h16* wt2  = (h16*)carve((size_t)256 * 512 * 2);   // [512][256]
  h16* wt3  = (h16*)carve((size_t)512 * 4096 * 2);  // [4096][512]
  h16* wf1  = (h16*)carve((size_t)64 * 128 * 2);    // [128][64]
  h16* wf2  = (h16*)carve((size_t)128 * 256 * 2);   // [256][128]
  h16* h1   = (h16*)carve((size_t)BATCH * 256 * 2);
  h16* h2   = (h16*)carve((size_t)BATCH * 512 * 2);
  h16* f1   = (h16*)carve((size_t)BATCH * 128 * 2);
  float* spatialF = (float*)carve((size_t)BATCH * 4096 * 4);
  float* freqloF  = (float*)carve((size_t)BATCH * 256 * 4);

  // Converts / transposes.
  k_cvt_f16<<<(BATCH * 64 + 255) / 256, 256, 0, stream>>>(msg, msgH, BATCH * 64);
  k_transpose_f16<<<(64 * 256 + 255) / 256, 256, 0, stream>>>(sW1, wt1, 64, 256);
  k_transpose_f16<<<(256 * 512 + 255) / 256, 256, 0, stream>>>(sW2, wt2, 256, 512);
  k_transpose_f16<<<(512 * 4096 + 255) / 256, 256, 0, stream>>>(sW3, wt3, 512, 4096);
  k_transpose_f16<<<(64 * 128 + 255) / 256, 256, 0, stream>>>(fW1, wf1, 64, 128);
  k_transpose_f16<<<(128 * 256 + 255) / 256, 256, 0, stream>>>(fW2, wf2, 128, 256);

  // GEMM chain (grid = (N/64, M/32), one wave per block).
  dim3 g1(256 / 64, BATCH / 32);
  k_gemm_wmma<<<g1, 32, 0, stream>>>(msgH, wt1, sb1, h1, nullptr, BATCH, 256, 64, 0);
  dim3 g2(512 / 64, BATCH / 32);
  k_gemm_wmma<<<g2, 32, 0, stream>>>(h1, wt2, sb2, h2, nullptr, BATCH, 512, 256, 0);
  dim3 g3(4096 / 64, BATCH / 32);
  k_gemm_wmma<<<g3, 32, 0, stream>>>(h2, wt3, sb3, nullptr, spatialF, BATCH, 4096, 512, 1);
  dim3 g4(128 / 64, BATCH / 32);
  k_gemm_wmma<<<g4, 32, 0, stream>>>(msgH, wf1, fb1, f1, nullptr, BATCH, 128, 64, 0);
  dim3 g5(256 / 64, BATCH / 32);
  k_gemm_wmma<<<g5, 32, 0, stream>>>(f1, wf2, fb2, nullptr, freqloF, BATCH, 256, 128, 1);

  // Assemble channels + Gaussian blur.
  k_assemble_blur<<<BATCH, 256, 0, stream>>>(spatialF, freqloF, msg, out);
}
